// GNN_infer_51539607552476
// MI455X (gfx1250) — matrix operand order repaired and checked by reading, not compiled, not run on hardware
//
#include <hip/hip_runtime.h>
#include <math.h>

// ---------------------------------------------------------------------------
// MI455X / gfx1250 graph-reasoning forward pass.
// All contractions run on V_WMMA_F32_16X16X32_F16. The im2col buffer is
// stored s-major in WMMA-A fragment order so each k-step A fragment is one
// contiguous 32B (v16h) load; B fragments are contiguous v16h LDS reads.
// ---------------------------------------------------------------------------

typedef __attribute__((ext_vector_type(16))) _Float16 v16h;
typedef __attribute__((ext_vector_type(8)))  float    v8f;

#define SS      9216            // H*W = 96*96
#define NBATCH  4
#define BSROWS  (NBATCH*SS)     // 36864 spatial rows
#define TILES   (BSROWS/16)     // 2304
#define GEMM_GRID (TILES/4)     // 4 waves (M-tiles) per 128-thread block
#define KPAD    192             // 180 rounded to 6 k-steps of 32

// fast sigmoid: v_exp_f32 + v_rcp_f32 (avoids the IEEE div expansion)
__device__ __forceinline__ float sigm(float x)
{
  return __builtin_amdgcn_rcpf(1.f + __expf(-x));
}

// A-fragment permutation: logical K -> storage index so that a[i] = buf[base+i]
// matches the 16-bit A-matrix VGPR layout (ISA 7.12.2).
__device__ __forceinline__ int aperm(int k)
{
  int q = k >> 5, r = k & 31;
  return (q << 5) | (((r >> 3) & 1) << 4) | (r & 7) | (((r >> 4) & 1) << 3);
}

// ---------------- grouped attention: sigmoid(einsum('bnchw,nc')+b) ----------
__global__ __launch_bounds__(256) void grouped_att_kernel(
    const float* __restrict__ nodes, const float* __restrict__ w,
    const float* __restrict__ bias, int N, float* __restrict__ out) // (N,B,S)
{
  int i = blockIdx.x * blockDim.x + threadIdx.x;
  if (i >= N * BSROWS) return;
  int s = i % SS; int nb = i / SS; int b = nb % NBATCH; int n = nb / NBATCH;
  float acc = bias[n];
#pragma unroll
  for (int c = 0; c < 10; ++c)
    acc += nodes[((size_t)(b * N + n) * 10 + c) * SS + s] * w[n * 10 + c];
  out[i] = sigm(acc);
}

// ---------------- decomp attention: sigmoid(w . [fea(256), xh(10), xl(10)]+b)
__global__ __launch_bounds__(256) void decomp_att_kernel(
    const float* __restrict__ fea,
    const float* __restrict__ xh, int bsh,
    const float* __restrict__ xl, int bsl,
    const float* __restrict__ w, const float* __restrict__ bias,
    float* __restrict__ out)      // plane (B,S)
{
  int i = blockIdx.x * blockDim.x + threadIdx.x;
  if (i >= BSROWS) return;
  int b = i / SS, s = i % SS;
  float acc = bias[0];
  const float* fb = fea + (size_t)b * 256 * SS + s;
  for (int c = 0; c < 256; ++c) acc += fb[(size_t)c * SS] * w[c];
#pragma unroll
  for (int c = 0; c < 10; ++c) acc += xh[(size_t)b * bsh + (size_t)c * SS + s] * w[256 + c];
#pragma unroll
  for (int c = 0; c < 10; ++c) acc += xl[(size_t)b * bsl + (size_t)c * SS + s] * w[266 + c];
  out[i] = sigm(acc);
}

// ---------------- pointwise (1x1 conv) WMMA GEMM ----------------------------
// MODE 0: relu(g*acc+b)   MODE 1: relu(x0 + g*acc + b)   (conv_update)
template<int CIN>
__device__ __forceinline__ float pw_fetch(
    int k, int b, int s,
    const float* i0, int b0, float v0, const float* i1, int b1, float v1,
    const float* i2, int b2, float v2, const float* i3, int b3, float v3)
{
  if (k >= CIN) return 0.f;                  // folds: k is constant after unroll
  const int j = k / 10, c = k - j * 10;      // folds
  const float* p; int bs; float sv;
  if (j == 0)      { p = i0; bs = b0; sv = v0; }
  else if (j == 1) { p = i1; bs = b1; sv = v1; }
  else if (j == 2) { p = i2; bs = b2; sv = v2; }
  else             { p = i3; bs = b3; sv = v3; }
  return p[(size_t)b * bs + (size_t)c * SS + s] * sv;
}

template<int CIN, int MODE>
__global__ __launch_bounds__(128) void pw_gemm_kernel(
    const float* __restrict__ in0, int bs0, const float* __restrict__ sc0,
    const float* __restrict__ in1, int bs1, const float* __restrict__ sc1,
    const float* __restrict__ in2, int bs2, const float* __restrict__ sc2,
    const float* __restrict__ in3, int bs3, const float* __restrict__ sc3,
    const float* __restrict__ W, const float* __restrict__ G,
    const float* __restrict__ Bt,
    float* __restrict__ out, int obs)
{
  __shared__ alignas(32) _Float16 wl[16 * 64];   // (n, k) f16, zero padded
  for (int i = threadIdx.x; i < 16 * 64; i += blockDim.x) {
    int n = i >> 6, k = i & 63;
    wl[i] = (_Float16)((n < 10 && k < CIN) ? W[n * CIN + k] : 0.f);
  }
  __syncthreads();

  int lane  = threadIdx.x & 31;
  int tile  = blockIdx.x * 4 + (threadIdx.x >> 5);
  int khalf = lane >> 4;
  int rowb  = tile * 16;
  int b     = rowb / SS;
  int s0    = rowb - b * SS;                 // tile never crosses a batch
  int s     = s0 + (lane & 15);
  int n     = lane & 15;

  float v0 = sc0 ? sc0[b * SS + s] : 1.f;
  float v1 = sc1 ? sc1[b * SS + s] : 1.f;
  float v2 = sc2 ? sc2[b * SS + s] : 1.f;
  float v3 = sc3 ? sc3[b * SS + s] : 1.f;

  v8f acc = {};
  constexpr int KSTEPS = (CIN + 31) / 32;
#pragma unroll
  for (int ks = 0; ks < KSTEPS; ++ks) {
    v16h a;
#pragma unroll
    for (int v = 0; v < 8; ++v) {
      int ka = (v * 2) + khalf * 16 + ks * 32; // A layout
      a[2 * v]     = (_Float16)pw_fetch<CIN>(ka,     b, s, in0, bs0, v0, in1, bs1, v1, in2, bs2, v2, in3, bs3, v3);
      a[2 * v + 1] = (_Float16)pw_fetch<CIN>(ka + 1, b, s, in0, bs0, v0, in1, bs1, v1, in2, bs2, v2, in3, bs3, v3);
    }
    v16h bf = *(const v16h*)&wl[n * 64 + khalf * 16 + ks * 32];       // B layout
    acc = __builtin_amdgcn_wmma_f32_16x16x32_f16(false, a, false, bf, (short)0, acc, false, false);
  }

  if (n < 10) {
    float g = G[n], be = Bt[n];
    float* ob = out + (size_t)b * obs + (size_t)n * SS + s0 + khalf * 8;
    const float* rb = in0 + (size_t)b * bs0 + (size_t)n * SS + s0 + khalf * 8;
#pragma unroll
    for (int r = 0; r < 8; ++r) {
      float val = g * acc[r] + be;
      if (MODE == 1) val += rb[r];
      ob[r] = fmaxf(val, 0.f);
    }
  }
}

// ---------------- im2col-GEMM (K=180 pad 192, 6 WMMA k-steps) ---------------
// col buffer layout: (B, S, 192) f16, K in A-fragment order (aperm).
// MODE 0: acc + bias[n], sigmoid for n >= 18 (offset conv)
// MODE 1: relu(g*acc+b)                 (dfconv stage 1)
// MODE 2: relu(g*acc+b)*(1-attA)*attB   (part_dep stage 2)
template<int NOUT, int MODE>
__global__ __launch_bounds__(128) void col_gemm_kernel(
    const _Float16* __restrict__ col,
    const float* __restrict__ W,        // (NOUT, 180), k = c*9+t
    const float* __restrict__ P0,       // bias or beta
    const float* __restrict__ P1,       // gamma or null
    const float* __restrict__ attA, const float* __restrict__ attB,
    float* __restrict__ out)            // (B, NOUT, S)
{
  constexpr int NPAD = (NOUT + 15) & ~15;
  __shared__ alignas(32) _Float16 wl[NPAD * KPAD];
  for (int i = threadIdx.x; i < NPAD * KPAD; i += blockDim.x) {
    int nn = i / KPAD, k = aperm(i - nn * KPAD);
    wl[i] = (_Float16)((nn < NOUT && k < 180) ? W[nn * 180 + k] : 0.f);
  }
  __syncthreads();

  int lane  = threadIdx.x & 31;
  int tile  = blockIdx.x * 4 + (threadIdx.x >> 5);
  int khalf = lane >> 4;
  int rowb  = tile * 16;
  int b     = rowb / SS;
  int s0    = rowb - b * SS;
  int s     = s0 + (lane & 15);
  int ng    = blockIdx.y * 16 + (lane & 15);

  const _Float16* arow = col + ((size_t)b * SS + s) * KPAD + khalf * 16;
  const _Float16* brow = &wl[ng * KPAD + khalf * 16];

  v8f acc = {};
#pragma unroll
  for (int ks = 0; ks < 6; ++ks) {
    if (ks < 5) __builtin_prefetch(arow + (ks + 1) * 32, 0, 1);
    v16h a  = *(const v16h*)(arow + ks * 32);   // fragment-order contiguous
    v16h bf = *(const v16h*)(brow + ks * 32);
    acc = __builtin_amdgcn_wmma_f32_16x16x32_f16(false, a, false, bf, (short)0, acc, false, false);
  }

  if (ng < NOUT) {
    float* ob = out + ((size_t)b * NOUT + ng) * SS + s0 + khalf * 8;
    const float* aap = (MODE == 2) ? attA + (size_t)b * SS + s0 + khalf * 8 : nullptr;
    const float* abp = (MODE == 2) ? attB + (size_t)b * SS + s0 + khalf * 8 : nullptr;
#pragma unroll
    for (int r = 0; r < 8; ++r) {
      float val = acc[r];
      if (MODE == 0) {
        val += P0[ng];
        if (ng >= 18) val = sigm(val);
      } else {
        val = fmaxf(P1[ng] * val + P0[ng], 0.f);
        if (MODE == 2) val *= (1.f - aap[r]) * abp[r];
      }
      ob[r] = val;
    }
  }
}

// ---------------- zero the K-pad (storage idx 160..191) of the col buffer ---
__global__ __launch_bounds__(256) void colpad_kernel(_Float16* __restrict__ col)
{
  int i = blockIdx.x * blockDim.x + threadIdx.x;
  if (i >= BSROWS) return;
  _Float16* p = col + (size_t)i * KPAD + 160;
#pragma unroll
  for (int k = 0; k < 32; ++k) p[k] = (_Float16)0.f;
}

// ---------------- im2col for the 3x3 offset conv ----------------------------
__global__ __launch_bounds__(256) void im2col_kernel(
    const float* __restrict__ p0, int bs0, int c0,
    const float* __restrict__ p1, int bs1,
    _Float16* __restrict__ col)
{
  int i = blockIdx.x * blockDim.x + threadIdx.x;
  if (i >= NBATCH * 9 * SS) return;
  int s = i % SS; int bt = i / SS; int t = bt % 9; int b = bt / 9;
  int y = s / 96, x = s - y * 96;
  int ys = y + t / 3 - 1, xs = x + t % 3 - 1;
  bool inb = (ys >= 0) && (ys < 96) && (xs >= 0) && (xs < 96);
  int sp = ys * 96 + xs;
  _Float16* cb = col + ((size_t)b * SS + s) * KPAD;
#pragma unroll
  for (int c = 0; c < 20; ++c) {
    float v = 0.f;
    if (inb) v = (c < c0) ? p0[(size_t)b * bs0 + (size_t)c * SS + sp]
                          : p1[(size_t)b * bs1 + (size_t)(c - c0) * SS + sp];
    cb[aperm(c * 9 + t)] = (_Float16)v;
  }
}

// ---------------- deformable bilinear sampling * mask -> f16 column ---------
__device__ __forceinline__ float tapf(const float* ch, int yy, int xx)
{
  return (yy >= 0 && yy < 96 && xx >= 0 && xx < 96) ? ch[yy * 96 + xx] : 0.f;
}

__global__ __launch_bounds__(256) void sample_kernel(
    const float* __restrict__ p0, int bs0, int c0,
    const float* __restrict__ p1, int bs1,
    const float* __restrict__ om,       // (B,27,S): dy[0:9], dx[9:18], m[18:27]
    _Float16* __restrict__ col)
{
  int i = blockIdx.x * blockDim.x + threadIdx.x;
  if (i >= NBATCH * 9 * SS) return;
  int s = i % SS; int bt = i / SS; int t = bt % 9; int b = bt / 9;
  int y = s / 96, x = s - y * 96;
  float dy = om[((size_t)b * 27 + t) * SS + s];
  float dx = om[((size_t)b * 27 + 9 + t) * SS + s];
  float mk = om[((size_t)b * 27 + 18 + t) * SS + s];
  float py = (float)(y + t / 3 - 1) + dy;
  float px = (float)(x + t % 3 - 1) + dx;
  float fy = floorf(py), fx = floorf(px);
  float wy = py - fy, wx = px - fx;
  int yi = (int)fy, xi = (int)fx;
  float w00 = (1.f - wy) * (1.f - wx), w01 = (1.f - wy) * wx;
  float w10 = wy * (1.f - wx),         w11 = wy * wx;
  _Float16* cb = col + ((size_t)b * SS + s) * KPAD;
#pragma unroll
  for (int c = 0; c < 20; ++c) {
    const float* ch = (c < c0) ? (p0 + (size_t)b * bs0 + (size_t)c * SS)
                               : (p1 + (size_t)b * bs1 + (size_t)(c - c0) * SS);
    float v = w00 * tapf(ch, yi, xi)     + w01 * tapf(ch, yi, xi + 1)
            + w10 * tapf(ch, yi + 1, xi) + w11 * tapf(ch, yi + 1, xi + 1);
    cb[aperm(c * 9 + t)] = (_Float16)(v * mk);
  }
}

// ---------------- segment max over incoming edges ---------------------------
__global__ __launch_bounds__(256) void segmax_kernel(
    const float* __restrict__ emsg, float* __restrict__ xpp)
{
  const int per = NBATCH * 10 * SS;
  int i = blockIdx.x * blockDim.x + threadIdx.x;
  if (i >= 6 * per) return;
  int seg = i / per; int r = i - seg * per;
  float m;
  if (seg == 0)      m = emsg[(size_t)1 * per + r];
  else if (seg == 1) m = fmaxf(fmaxf(emsg[(size_t)0 * per + r],
                                     emsg[(size_t)4 * per + r]),
                                     emsg[(size_t)7 * per + r]);
  else if (seg == 2) m = fmaxf(emsg[(size_t)2 * per + r], emsg[(size_t)6 * per + r]);
  else if (seg == 3) m = emsg[(size_t)5 * per + r];
  else if (seg == 4) m = fmaxf(emsg[(size_t)3 * per + r], emsg[(size_t)9 * per + r]);
  else               m = emsg[(size_t)8 * per + r];
  xpp[i] = m;
}

// ---------------- final attention combine -----------------------------------
__global__ __launch_bounds__(256) void att_combine_kernel(
    const float* __restrict__ attP,   // (8,B,S): hp_atts[0..5], att_u, att_l
    const float* __restrict__ patt,   // (6,B,S)
    const float* __restrict__ hatt,   // (2,B,S)
    float* __restrict__ out)          // (B,8,S)
{
  int i = blockIdx.x * blockDim.x + threadIdx.x;
  if (i >= NBATCH * 8 * SS) return;
  int s = i % SS; int cb = i / SS; int ch = cb % 8; int b = cb / 8;
  float f1 = attP[((size_t)ch * NBATCH + b) * SS + s];
  float f2 = (ch < 6) ? patt[((size_t)ch * NBATCH + b) * SS + s]
                      : hatt[((size_t)(ch - 6) * NBATCH + b) * SS + s];
  out[((size_t)b * 8 + ch) * SS + s] = 0.5f * (f1 + f2);
}

// ---------------------------------------------------------------------------
// Host orchestration helpers
// ---------------------------------------------------------------------------
template<int CIN, int MODE>
static void run_pw(hipStream_t st,
    const float* i0, int b0, const float* s0,
    const float* i1, int b1, const float* s1,
    const float* i2, int b2, const float* s2,
    const float* i3, int b3, const float* s3,
    const float* W, const float* G, const float* Bt,
    float* out, int obs)
{
  pw_gemm_kernel<CIN, MODE><<<GEMM_GRID, 128, 0, st>>>(
      i0, b0, s0, i1, b1, s1, i2, b2, s2, i3, b3, s3, W, G, Bt, out, obs);
}

template<int NOUT, int MODE>
static void run_dfconv(hipStream_t st,
    const float* p0, int bs0, int c0, const float* p1, int bs1,
    const float* offw, const float* offb,
    const float* w, const float* g, const float* be,
    const float* attA, const float* attB,
    _Float16* colA, float* om, float* out)
{
  im2col_kernel<<<1296, 256, 0, st>>>(p0, bs0, c0, p1, bs1, colA);
  col_gemm_kernel<27, 0><<<dim3(GEMM_GRID, 2), 128, 0, st>>>(
      colA, offw, offb, nullptr, nullptr, nullptr, om);
  sample_kernel<<<1296, 256, 0, st>>>(p0, bs0, c0, p1, bs1, om, colA);
  col_gemm_kernel<NOUT, MODE><<<dim3(GEMM_GRID, (NOUT + 15) / 16), 128, 0, st>>>(
      colA, w, be, g, attA, attB, out);
}

static void run_part_dep(hipStream_t st,
    const float* pB, int bsB, const float* pA, int bsA,
    const float* o1w, const float* o1b, const float* w1, const float* g1, const float* b1,
    const float* o2w, const float* o2b, const float* w2, const float* g2, const float* b2,
    const float* attA, const float* attB,
    _Float16* colA, float* om, float* s1buf, float* out)
{
  // x = concat([pB, pA]); stage1: 20ch bn_relu; stage2: 10ch bn_relu*(1-attA)*attB
  run_dfconv<20, 1>(st, pB, bsB, 10, pA, bsA, o1w, o1b, w1, g1, b1,
                    nullptr, nullptr, colA, om, s1buf);
  run_dfconv<10, 2>(st, s1buf, 20 * SS, 20, nullptr, 0, o2w, o2b, w2, g2, b2,
                    attA, attB, colA, om, out);
}

extern "C" void kernel_launch(void* const* d_in, const int* in_sizes, int n_in,
                              void* d_out, int out_size, void* d_ws, size_t ws_size,
                              hipStream_t stream)
{
  (void)in_sizes; (void)n_in; (void)out_size; (void)ws_size;
  const float* xp   = (const float*)d_in[0];   // (B,6,10,S)
  const float* xh   = (const float*)d_in[1];   // (B,2,10,S)
  const float* xf   = (const float*)d_in[2];   // (B,10,S)
  const float* pfea = (const float*)d_in[4];   // (B,256,S)
  const float* hfea = (const float*)d_in[5];
  // params (flattened dict order)
  const float* fc_w = (const float*)d_in[7];  const float* fc_g = (const float*)d_in[8];  const float* fc_b = (const float*)d_in[9];
  const float* fu_w = (const float*)d_in[10]; const float* fu_g = (const float*)d_in[11]; const float* fu_b = (const float*)d_in[12];
  const float* ha_w = (const float*)d_in[13]; const float* ha_b = (const float*)d_in[14];
  const float* hcu_w = (const float*)d_in[15]; const float* hcu_g = (const float*)d_in[16]; const float* hcu_b = (const float*)d_in[17];
  const float* hcl_w = (const float*)d_in[18]; const float* hcl_g = (const float*)d_in[19]; const float* hcl_b = (const float*)d_in[20];
  const float* dph[10]; for (int i = 0; i < 10; ++i) dph[i] = (const float*)d_in[21 + i];
  const float* dpl[10]; for (int i = 0; i < 10; ++i) dpl[i] = (const float*)d_in[31 + i];
  const float* dcu_w = (const float*)d_in[41]; const float* dcu_b = (const float*)d_in[42];
  const float* dcl_w = (const float*)d_in[43]; const float* dcl_b = (const float*)d_in[44];
  const float* huu_w = (const float*)d_in[45]; const float* huu_g = (const float*)d_in[46]; const float* huu_b = (const float*)d_in[47];
  const float* hul_w = (const float*)d_in[48]; const float* hul_g = (const float*)d_in[49]; const float* hul_b = (const float*)d_in[50];
  const float* pa_w = (const float*)d_in[51]; const float* pa_b = (const float*)d_in[52];
  const float* pdc_w = (const float*)d_in[53]; const float* pdc_b = (const float*)d_in[54];
  const float* edp[10]; for (int i = 0; i < 10; ++i) edp[i] = (const float*)d_in[55 + i];
  const float* pu_w = (const float*)d_in[65]; const float* pu_g = (const float*)d_in[66]; const float* pu_b = (const float*)d_in[67];

  const int BSn = NBATCH * SS;
  const float* xh0 = xh;                 const int xh_bs = 20 * SS;
  const float* xh1 = xh + 10 * SS;
  const int xp_bs = 60 * SS;
  const int t_bs  = 10 * SS;

  float* out_xp  = (float*)d_out;                       // (B,6,10,S)
  float* out_xh  = out_xp + (size_t)NBATCH * 60 * SS;   // (B,2,10,S)
  float* out_xf  = out_xh + (size_t)NBATCH * 20 * SS;   // (B,10,S)
  float* out_att = out_xf + (size_t)NBATCH * 10 * SS;   // (B,8,S)

  char* w8 = (char*)d_ws; size_t woff = 0;
  auto wsalloc = [&](size_t bytes) -> void* {
    void* p = w8 + woff; woff = (woff + bytes + 255) & ~(size_t)255; return p;
  };
  float* fmsg = (float*)wsalloc((size_t)10 * BSn * 4);
  float* hatt = (float*)wsalloc((size_t)2  * BSn * 4);   // (2,B,S): a_u, a_l
  float* patt = (float*)wsalloc((size_t)6  * BSn * 4);   // (6,B,S)
  float* attP = (float*)wsalloc((size_t)8  * BSn * 4);   // hp 0..5, att_u, att_l
  float* xphu = (float*)wsalloc((size_t)10 * BSn * 4);
  float* xphl = (float*)wsalloc((size_t)10 * BSn * 4);
  float* xlh  = (float*)wsalloc((size_t)10 * BSn * 4);
  float* xuh  = (float*)wsalloc((size_t)10 * BSn * 4);
  float* om   = (float*)wsalloc((size_t)27 * BSn * 4);
  float* s1b  = (float*)wsalloc((size_t)20 * BSn * 4);
  _Float16* colA = (_Float16*)wsalloc((size_t)BSn * KPAD * 2);
  float* emsg = (float*)wsalloc((size_t)100 * BSn * 4);  // (10 edges, B,10,S)
  float* xpp  = (float*)wsalloc((size_t)60  * BSn * 4);  // (6, B,10,S)

  // 0) zero the K-pad region of the column buffer (persists across im2col)
  colpad_kernel<<<BSROWS / 256, 256, 0, stream>>>(colA);

  // 1) grouped attentions
  grouped_att_kernel<<<(2 * BSn) / 256, 256, 0, stream>>>(xh, ha_w, ha_b, 2, hatt);
  grouped_att_kernel<<<(6 * BSn) / 256, 256, 0, stream>>>(xp, pa_w, pa_b, 6, patt);
  const float* a_u = hatt;            // plane 0
  const float* a_l = hatt + BSn;      // plane 1

  // 2) full node
  run_pw<20, 0>(stream, xh0, xh_bs, nullptr, xh1, xh_bs, nullptr,
                nullptr, 0, nullptr, nullptr, 0, nullptr,
                fc_w, fc_g, fc_b, fmsg, t_bs);
  run_pw<20, 1>(stream, xf, t_bs, nullptr, fmsg, t_bs, nullptr,
                nullptr, 0, nullptr, nullptr, 0, nullptr,
                fu_w, fu_g, fu_b, out_xf, t_bs);

  // 3) compositions
  run_pw<40, 0>(stream, xp, xp_bs, nullptr, xp + 10 * SS, xp_bs, nullptr,
                xp + 20 * SS, xp_bs, nullptr, xp + 30 * SS, xp_bs, nullptr,
                hcu_w, hcu_g, hcu_b, xphu, t_bs);
  run_pw<20, 0>(stream, xp + 40 * SS, xp_bs, nullptr, xp + 50 * SS, xp_bs, nullptr,
                nullptr, 0, nullptr, nullptr, 0, nullptr,
                hcl_w, hcl_g, hcl_b, xphl, t_bs);

  // 4) half part-deps
  run_part_dep(stream, xh0, xh_bs, xh1, xh_bs,
               dph[0], dph[1], dph[2], dph[3], dph[4], dph[5], dph[6], dph[7], dph[8], dph[9],
               a_l, a_u, colA, om, s1b, xlh);
  run_part_dep(stream, xh1, xh_bs, xh0, xh_bs,
               dpl[0], dpl[1], dpl[2], dpl[3], dpl[4], dpl[5], dpl[6], dpl[7], dpl[8], dpl[9],
               a_u, a_l, colA, om, s1b, xuh);

  // 5) decomposition attentions
  for (int i = 0; i < 6; ++i) {
    const float* xhi = (i < 4) ? xh0 : xh1;
    decomp_att_kernel<<<BSn / 256, 256, 0, stream>>>(
        pfea, xhi, xh_bs, xp + i * 10 * SS, xp_bs,
        pdc_w + i * 276, pdc_b + i, attP + (size_t)i * BSn);
  }
  decomp_att_kernel<<<BSn / 256, 256, 0, stream>>>(hfea, xf, t_bs, xh0, xh_bs,
                                                   dcu_w, dcu_b, attP + (size_t)6 * BSn);
  decomp_att_kernel<<<BSn / 256, 256, 0, stream>>>(hfea, xf, t_bs, xh1, xh_bs,
                                                   dcl_w, dcl_b, attP + (size_t)7 * BSn);

  // 6) half updates
  run_pw<40, 1>(stream, xh0, xh_bs, nullptr, xphu, t_bs, nullptr, xlh, t_bs, nullptr,
                xf, t_bs, attP + (size_t)6 * BSn,
                huu_w, huu_g, huu_b, out_xh, 20 * SS);
  run_pw<40, 1>(stream, xh1, xh_bs, nullptr, xphl, t_bs, nullptr, xuh, t_bs, nullptr,
                xf, t_bs, attP + (size_t)7 * BSn,
                hul_w, hul_g, hul_b, out_xh + 10 * SS, 20 * SS);

  // 7) edge part-deps
  static const int ESRC[10] = {0, 1, 1, 1, 2, 2, 3, 4, 4, 5};
  static const int EDST[10] = {1, 0, 2, 4, 1, 3, 2, 1, 5, 4};
  for (int e = 0; e < 10; ++e) {
    const float* pA = xp + (size_t)ESRC[e] * 10 * SS;
    const float* pB = xp + (size_t)EDST[e] * 10 * SS;
    run_part_dep(stream, pB, xp_bs, pA, xp_bs,
                 edp[0] + (size_t)e * 27 * 180, edp[1] + (size_t)e * 27,
                 edp[2] + (size_t)e * 20 * 180, edp[3] + (size_t)e * 20, edp[4] + (size_t)e * 20,
                 edp[5] + (size_t)e * 27 * 180, edp[6] + (size_t)e * 27,
                 edp[7] + (size_t)e * 10 * 180, edp[8] + (size_t)e * 10, edp[9] + (size_t)e * 10,
                 patt + (size_t)ESRC[e] * BSn, patt + (size_t)EDST[e] * BSn,
                 colA, om, s1b, emsg + (size_t)e * 10 * BSn);
  }

  // 8) segment max over incoming edges -> xpp
  segmax_kernel<<<(6 * 10 * BSn) / 256, 256, 0, stream>>>(emsg, xpp);

  // 9) part updates
  for (int i = 0; i < 6; ++i) {
    const float* xhi = (i < 4) ? xh0 : xh1;
    run_pw<30, 1>(stream, xp + (size_t)i * 10 * SS, xp_bs, nullptr,
                  xpp + (size_t)i * 10 * BSn, t_bs, nullptr,
                  xhi, xh_bs, attP + (size_t)i * BSn,
                  nullptr, 0, nullptr,
                  pu_w + i * 300, pu_g + i * 10, pu_b + i * 10,
                  out_xp + (size_t)i * 10 * SS, 60 * SS);
  }

  // 10) final attention output
  att_combine_kernel<<<(8 * BSn) / 256, 256, 0, stream>>>(attP, patt, hatt, out_att);
}
